// CustomAttention_31318901522898
// MI455X (gfx1250) — matrix operand order repaired and checked by reading, not compiled, not run on hardware
//
#include <hip/hip_runtime.h>
#include <hip/hip_bf16.h>
#include <math.h>

// ---------------------------------------------------------------------------
// Problem constants (from reference): B=4, F=T=2048, D=768, N=12, H=64
// ---------------------------------------------------------------------------
constexpr int B_ = 4;
constexpr int F_ = 2048;
constexpr int T_ = 2048;
constexpr int D_ = 768;
constexpr int N_ = 12;
constexpr int H_ = 64;
constexpr int NH_ = N_ * H_;          // 768
constexpr float SCALE = 0.125f;       // 1/sqrt(64)

typedef __bf16 bf16_t;
typedef __attribute__((ext_vector_type(16))) __bf16 v16bf;
typedef __attribute__((ext_vector_type(8)))  __bf16 v8bf;
typedef __attribute__((ext_vector_type(8)))  float  v8f;

// may_alias views for reinterpret-cast LDS/global accesses
typedef v8bf  __attribute__((may_alias)) v8bf_a;
typedef uint4 __attribute__((may_alias)) uint4_a;
typedef float4 __attribute__((may_alias)) float4_a;

__device__ __forceinline__ v16bf join16(v8bf lo, v8bf hi) {
    return __builtin_shufflevector(lo, hi, 0,1,2,3,4,5,6,7,8,9,10,11,12,13,14,15);
}

__device__ __forceinline__ v8f wmma_bf16(v16bf a, v16bf b, v8f c) {
    return __builtin_amdgcn_wmma_f32_16x16x32_bf16(
        /*neg_a=*/false, a, /*neg_b=*/false, b,
        /*c_mod=*/(short)0, c, /*reuse_a=*/false, /*reuse_b=*/false);
}

// Wait for all outstanding async global<->LDS transfers (ASYNCcnt == 0).
__device__ __forceinline__ void wait_async0() {
#if __has_builtin(__builtin_amdgcn_s_wait_asynccnt)
    __builtin_amdgcn_s_wait_asynccnt(0);
#else
    asm volatile("s_wait_asynccnt 0x0" ::: "memory");
#endif
}

// Async DMA of 32 contiguous bytes global -> LDS (two B128 ops; the
// instruction's IOFFSET applies to both the LDS and the global address).
__device__ __forceinline__ void async_copy32(const bf16_t* gsrc, bf16_t* lds_dst) {
    unsigned lds = (unsigned)(unsigned long long)lds_dst;   // addr[31:0] = LDS offset
    unsigned long long ga = (unsigned long long)gsrc;
    asm volatile(
        "global_load_async_to_lds_b128 %0, %1, off\n\t"
        "global_load_async_to_lds_b128 %0, %1, off offset:16"
        :: "v"(lds), "v"(ga) : "memory");
}

// ---------------------------------------------------------------------------
// Kernel 1: QKV projection GEMM.  C[m, c] = A[m, :] * W[:, c] + bias[c]
//   mode (blockIdx.z): 0 -> Q (A=from),  1 -> K (A=to),  2 -> V (A=to)
//   Q/K written as bf16 [B][N][rows][H];  V written transposed [B][N][H][T].
// Block = 128 threads (4 waves); tile = 64 rows x 64 cols (one head).
// (fp32 -> bf16 conversion happens in-flight, so staging stays on VALU path)
// ---------------------------------------------------------------------------
__global__ __launch_bounds__(128)
void qkv_kernel(const float* __restrict__ from_t, const float* __restrict__ to_t,
                const float* __restrict__ Wq, const float* __restrict__ Wk,
                const float* __restrict__ Wv,
                const float* __restrict__ bq, const float* __restrict__ bk,
                const float* __restrict__ bv,
                bf16_t* __restrict__ Qws, bf16_t* __restrict__ Kws,
                bf16_t* __restrict__ VwsT)
{
    const int mode = blockIdx.z;
    const float* Ain  = (mode == 0) ? from_t : to_t;
    const float* W    = (mode == 0) ? Wq : (mode == 1 ? Wk : Wv);
    const float* bias = (mode == 0) ? bq : (mode == 1 ? bk : bv);

    const int m0   = blockIdx.x * 64;   // row tile  (rows = B*F flattened)
    const int head = blockIdx.y;        // column tile == one head (64 cols)

    __shared__ bf16_t As [64][40];      // [m][k]  stride 40 (80B, 16B-mult)
    __shared__ bf16_t WsT[64][40];      // [c][k]  (weights transposed)

    const int tid  = threadIdx.x;
    const int lane = tid & 31;
    const int wave = tid >> 5;
    const int mln  = lane & 15;         // M (A/C) or N (B) index
    const int hi   = lane >> 4;
    const int khalf = hi * 8;           // A-frag K grouping
    const int kq    = hi * 16;          // unused (B frags use khalf too)

    v8f acc[4];
    #pragma unroll
    for (int i = 0; i < 4; ++i) acc[i] = (v8f){};

    for (int kk = 0; kk < D_; kk += 32) {
        // --- stage A tile: 64x32 fp32 -> bf16 LDS.  16 floats per thread.
        {
            const int row = tid >> 1;
            const int ch  = (tid & 1) * 16;
            const float* src = Ain + (size_t)(m0 + row) * D_ + kk + ch;
            #pragma unroll
            for (int j = 0; j < 16; j += 4) {
                float4 f = *(const float4_a*)(src + j);
                As[row][ch + j + 0] = (bf16_t)f.x;
                As[row][ch + j + 1] = (bf16_t)f.y;
                As[row][ch + j + 2] = (bf16_t)f.z;
                As[row][ch + j + 3] = (bf16_t)f.w;
            }
        }
        // --- stage W tile: 32(k) x 64(c), transposed into WsT[c][k].
        {
            const int k  = tid >> 2;
            const int cq = (tid & 3) * 16;
            const float* src = W + (size_t)(kk + k) * NH_ + head * 64 + cq;
            #pragma unroll
            for (int j = 0; j < 16; j += 4) {
                float4 f = *(const float4_a*)(src + j);
                WsT[cq + j + 0][k] = (bf16_t)f.x;
                WsT[cq + j + 1][k] = (bf16_t)f.y;
                WsT[cq + j + 2][k] = (bf16_t)f.z;
                WsT[cq + j + 3][k] = (bf16_t)f.w;
            }
        }
        __syncthreads();

        // A fragment: lane m = mln, K packed per ISA 16-bit A layout.
        v8bf alo = *(const v8bf_a*)&As[wave * 16 + mln][khalf];
        v8bf ahi = *(const v8bf_a*)&As[wave * 16 + mln][khalf + 16];
        v16bf afrag = join16(alo, ahi);

        #pragma unroll
        for (int nt = 0; nt < 4; ++nt) {
            v8bf blo = *(const v8bf_a*)&WsT[nt * 16 + mln][khalf];
            v8bf bhi = *(const v8bf_a*)&WsT[nt * 16 + mln][khalf + 16];
            acc[nt] = wmma_bf16(afrag, join16(blo, bhi), acc[nt]);
        }
        __syncthreads();
    }

    // --- epilogue: bias + store bf16 workspace
    const int rowbase = m0 + wave * 16 + 8 * hi;
    #pragma unroll
    for (int nt = 0; nt < 4; ++nt) {
        const int h = nt * 16 + mln;                 // column within head
        const float bb = bias[head * H_ + h];
        #pragma unroll
        for (int r = 0; r < 8; ++r) {
            const int row = rowbase + r;             // global row in [0, B*F)
            const int b = row / F_;
            const int f = row % F_;                  // F_ == T_
            const float val = acc[nt][r] + bb;
            if (mode == 0)
                Qws[(((size_t)b * N_ + head) * F_ + f) * H_ + h] = (bf16_t)val;
            else if (mode == 1)
                Kws[(((size_t)b * N_ + head) * T_ + f) * H_ + h] = (bf16_t)val;
            else
                VwsT[(((size_t)b * N_ + head) * H_ + h) * T_ + f] = (bf16_t)val;
        }
    }
}

// ---------------------------------------------------------------------------
// Kernel 2: fused flash attention per (b, n, 64-row f block).
//   - K/V tiles double-buffered in LDS, staged by async global->LDS DMA
//     (ASYNCcnt path), overlapping next-tile DMA with current-tile WMMA.
//   - S = (Q K^T) * scale + (1-mask)*-1e4, written to scores (fp32, once)
//   - online softmax (row max/sum via width-16 shuffles)
//   - O += P V  (P round-trips per-wave LDS to become an A fragment)
//   - context[b][f][n][h] = O / l
// Block = 128 threads (4 waves x 16 rows = 64 f-rows); t processed 32/iter.
// ---------------------------------------------------------------------------
__global__ __launch_bounds__(128)
void attn_kernel(const bf16_t* __restrict__ Qws, const bf16_t* __restrict__ Kws,
                 const bf16_t* __restrict__ VwsT, const int* __restrict__ mask,
                 float* __restrict__ ctx_out, float* __restrict__ scores_out)
{
    const int f0 = blockIdx.x * 64;
    const int n  = blockIdx.y;
    const int b  = blockIdx.z;

    __shared__ bf16_t Qs [64][72];         // [f][h]   144B rows (16B-mult)
    __shared__ bf16_t Ks [2][32][72];      // double-buffered [t][h]
    __shared__ bf16_t VsT[2][64][40];      // double-buffered [h][t]
    __shared__ bf16_t Ps [4][16][40];      // per-wave P tile [f][t]

    const int tid  = threadIdx.x;
    const int lane = tid & 31;
    const int wave = tid >> 5;
    const int mln  = lane & 15;
    const int hi   = lane >> 4;
    const int khalf = hi * 8;
    const int kq    = hi * 16;

    const bf16_t* Qbase = Qws  + (((size_t)b * N_ + n) * F_ + f0) * H_;
    const bf16_t* Kbase = Kws  + (((size_t)b * N_ + n) * T_) * H_;
    const bf16_t* Vbase = VwsT + (((size_t)b * N_ + n) * H_) * T_;
    const int*    Mbase = mask + (size_t)b * F_ * T_;
    float* Sbase = scores_out + (((size_t)b * N_ + n) * F_) * T_;

    // Per-thread source coordinates for the cooperative K/V tile DMA.
    const int krow = tid >> 2;             // 0..31  (t within tile)
    const int kch  = (tid & 3) * 16;       // 0..48  (h chunk)
    const int vh   = tid >> 1;             // 0..63  (h row)
    const int vch  = (tid & 1) * 16;       // 0/16   (t chunk)

    auto issue_tile = [&](int t0, int buf) {
        async_copy32(Kbase + (size_t)(t0 + krow) * H_ + kch, &Ks[buf][krow][kch]);
        async_copy32(Vbase + (size_t)vh * T_ + t0 + vch,     &VsT[buf][vh][vch]);
    };

    // Kick off DMA for tile 0, then stage Q synchronously (overlapped).
    issue_tile(0, 0);
    {
        const int row = tid >> 1;
        const int ch  = (tid & 1) * 32;
        const bf16_t* src = Qbase + (size_t)row * H_ + ch;
        #pragma unroll
        for (int j = 0; j < 32; j += 8)
            *(uint4_a*)&Qs[row][ch + j] = *(const uint4_a*)(src + j);
    }
    __syncthreads();

    // Q A-fragments (K-dim = H = 64 -> two 32-wide k-steps), loop-invariant.
    v16bf aq[2];
    #pragma unroll
    for (int ks = 0; ks < 2; ++ks) {
        v8bf lo = *(const v8bf_a*)&Qs[wave * 16 + mln][ks * 32 + khalf];
        v8bf hh = *(const v8bf_a*)&Qs[wave * 16 + mln][ks * 32 + khalf + 16];
        aq[ks] = join16(lo, hh);
    }

    v8f o[4];
    #pragma unroll
    for (int i = 0; i < 4; ++i) o[i] = (v8f){};
    float mrow[8], lrow[8];
    #pragma unroll
    for (int r = 0; r < 8; ++r) { mrow[r] = -1e30f; lrow[r] = 0.0f; }

    const int frow_base = f0 + wave * 16 + 8 * hi;

    for (int it = 0; it < T_ / 32; ++it) {
        const int t0  = it * 32;
        const int buf = it & 1;

        // Drain this tile's DMA, make it visible to all waves, then launch
        // the next tile's DMA into the alternate buffer (overlaps compute).
        wait_async0();
        __syncthreads();
        if (t0 + 32 < T_) issue_tile(t0 + 32, buf ^ 1);

        // --- S = Q K^T  (two 16x16 f32 tiles covering 32 t-columns)
        v8f s[2];
        #pragma unroll
        for (int tt = 0; tt < 2; ++tt) {
            v8f c = (v8f){};
            #pragma unroll
            for (int ks = 0; ks < 2; ++ks) {
                v8bf blo = *(const v8bf_a*)&Ks[buf][tt * 16 + mln][ks * 32 + khalf];
                v8bf bhi = *(const v8bf_a*)&Ks[buf][tt * 16 + mln][ks * 32 + khalf + 16];
                c = wmma_bf16(aq[ks], join16(blo, bhi), c);
            }
            s[tt] = c;
        }

        // --- scale + mask + scores write + online softmax, per row r
        #pragma unroll
        for (int r = 0; r < 8; ++r) {
            const int frow = frow_base + r;
            const int* mrp = Mbase + (size_t)frow * T_ + t0;
            float s0 = s[0][r] * SCALE + (float)(1 - mrp[mln])      * -10000.0f;
            float s1 = s[1][r] * SCALE + (float)(1 - mrp[16 + mln]) * -10000.0f;
            float* srp = Sbase + (size_t)frow * T_ + t0;
            srp[mln]      = s0;
            srp[16 + mln] = s1;

            float mx = fmaxf(s0, s1);
            #pragma unroll
            for (int off = 1; off < 16; off <<= 1)
                mx = fmaxf(mx, __shfl_xor(mx, off, 16));
            const float mn    = fmaxf(mrow[r], mx);
            const float alpha = __expf(mrow[r] - mn);
            const float p0 = __expf(s0 - mn);
            const float p1 = __expf(s1 - mn);
            float sum = p0 + p1;
            #pragma unroll
            for (int off = 1; off < 16; off <<= 1)
                sum += __shfl_xor(sum, off, 16);
            lrow[r] = lrow[r] * alpha + sum;
            mrow[r] = mn;
            #pragma unroll
            for (int nt = 0; nt < 4; ++nt) o[nt][r] *= alpha;

            Ps[wave][r + 8 * hi][mln]      = (bf16_t)p0;
            Ps[wave][r + 8 * hi][16 + mln] = (bf16_t)p1;
        }

        // --- O += P V   (A frag from per-wave Ps, B frags from VsT)
        v8bf plo = *(const v8bf_a*)&Ps[wave][mln][khalf];
        v8bf phi = *(const v8bf_a*)&Ps[wave][mln][khalf + 16];
        v16bf ap = join16(plo, phi);
        #pragma unroll
        for (int nt = 0; nt < 4; ++nt) {
            v8bf blo = *(const v8bf_a*)&VsT[buf][nt * 16 + mln][khalf];
            v8bf bhi = *(const v8bf_a*)&VsT[buf][nt * 16 + mln][khalf + 16];
            o[nt] = wmma_bf16(ap, join16(blo, bhi), o[nt]);
        }
    }

    // --- normalize and store context [b][f][n][h] fp32
    #pragma unroll
    for (int nt = 0; nt < 4; ++nt) {
        const int h = nt * 16 + mln;
        #pragma unroll
        for (int r = 0; r < 8; ++r) {
            const int frow = frow_base + r;
            ctx_out[(((size_t)b * F_ + frow) * N_ + n) * H_ + h] =
                o[nt][r] / lrow[r];
        }
    }
}

// ---------------------------------------------------------------------------
// Host launcher
// ---------------------------------------------------------------------------
extern "C" void kernel_launch(void* const* d_in, const int* in_sizes, int n_in,
                              void* d_out, int out_size, void* d_ws, size_t ws_size,
                              hipStream_t stream) {
    (void)in_sizes; (void)n_in; (void)out_size;

    const float* from_t = (const float*)d_in[0];
    const float* to_t   = (const float*)d_in[1];
    const int*   mask   = (const int*)  d_in[2];
    const float* Wq     = (const float*)d_in[3];
    const float* Wk     = (const float*)d_in[4];
    const float* Wv     = (const float*)d_in[5];
    const float* bq     = (const float*)d_in[6];
    const float* bk     = (const float*)d_in[7];
    const float* bv     = (const float*)d_in[8];

    float* ctx    = (float*)d_out;                              // [B,F,N,H]
    float* scores = ctx + (size_t)B_ * F_ * N_ * H_;            // [B,N,F,T]

    const size_t qkv_elems = (size_t)B_ * N_ * F_ * H_;         // 6,291,456
    bf16_t* Qws  = (bf16_t*)d_ws;
    bf16_t* Kws  = Qws + qkv_elems;
    bf16_t* VwsT = Kws + qkv_elems;
    if (ws_size < 3 * qkv_elems * sizeof(bf16_t)) return;

    dim3 g1((B_ * F_) / 64, N_, 3);     // (128, 12, 3)
    qkv_kernel<<<g1, 128, 0, stream>>>(from_t, to_t, Wq, Wk, Wv,
                                       bq, bk, bv, Qws, Kws, VwsT);

    dim3 g2(F_ / 64, N_, B_);           // (32, 12, 4)
    attn_kernel<<<g2, 128, 0, stream>>>(Qws, Kws, VwsT, mask, ctx, scores);
}
